// ExtSummModel_22222160790112
// MI455X (gfx1250) — matrix operand order repaired and checked by reading, not compiled, not run on hardware
//
#include <hip/hip_runtime.h>
#include <hip/hip_bf16.h>
#include <math.h>

typedef _Float16 hvec16 __attribute__((ext_vector_type(16)));
typedef float    fvec8  __attribute__((ext_vector_type(8)));

#define Bv   32
#define Sv   256
#define Wv   32
#define Ev   300
#define Hv   128
#define Tv   16
#define Mrows 8192            // B*S
#define KPAD 320              // E padded to mult of 32

// ---------------- WMMA fragment helpers (contiguous k = 16*(lane/16) + j order) ----------------

// A: 16x32 f16 tile, element [m,k], A row-major with leading dim lda.
// Lane layout: lanes 0-15 rows, elements 0-15 -> K 0..15 (+kb), kb = 16*(lane/16).
// Each lane reads one contiguous 32B run -> compiler merges to b128 loads.
__device__ __forceinline__ hvec16 load_a_frag(const _Float16* A, int lda) {
    int lane = threadIdx.x & 31;
    int m  = lane & 15;
    int kb = (lane >> 4) << 4;
    const _Float16* p = A + m * lda;
    hvec16 a;
#pragma unroll
    for (int j = 0; j < 16; ++j) {
        int k = kb + j;
        a[j] = p[k];
    }
    return a;
}

// B matrices are pre-swizzled to fragment-native order: element (k,n) of a K x N
// matrix lives at bswz(k,n): [ktile][ntile][lane][j], lane = ((k>>4)&1)*16 + (n&15),
// j = k&15.  A lane's fragment is one contiguous 32-byte block (2x global_load_b128),
// a wave's fragment is 1KB contiguous.
__host__ __device__ __forceinline__ size_t bswz(int k, int n, int ntiles) {
    return ((((size_t)(k >> 5) * ntiles + (n >> 4)) * 32
             + (size_t)(((k >> 4) & 1) * 16 + (n & 15))) << 4) + (k & 15);
}

__device__ __forceinline__ hvec16 load_b_frag_sw(const _Float16* Bsw, int kt, int nt, int ntiles) {
    int lane = threadIdx.x & 31;
    const hvec16* p = (const hvec16*)(Bsw + ((((size_t)kt * ntiles + nt) * 32 + lane) << 4));
    return *p;
}

// ---------------- weight prep (f32 -> f16 swizzled B layouts) ----------------

__global__ void conv_giB(const float* __restrict__ wf, const float* __restrict__ wb,
                         _Float16* __restrict__ out) {           // [320 x 768] swizzled
    int idx = blockIdx.x * 256 + threadIdx.x;
    if (idx >= KPAD * 768) return;
    int k = idx / 768, n = idx % 768;
    float v = 0.f;
    if (k < Ev) v = (n < 384) ? wf[n * Ev + k] : wb[(n - 384) * Ev + k];
    out[bswz(k, n, 48)] = (_Float16)v;
}

__global__ void conv_whhT(const float* __restrict__ wf, const float* __restrict__ wb,
                          _Float16* __restrict__ out) {          // [128 x 768] swizzled
    int idx = blockIdx.x * 256 + threadIdx.x;
    if (idx >= Hv * 768) return;
    int k = idx / 768, n = idx % 768;
    float v = (n < 384) ? wf[n * Hv + k] : wb[(n - 384) * Hv + k];
    out[bswz(k, n, 48)] = (_Float16)v;
}

__global__ void conv_watt(const float* __restrict__ in, _Float16* __restrict__ out) { // [512 x 512] swizzled
    int idx = blockIdx.x * 256 + threadIdx.x;
    if (idx >= 512 * 512) return;
    int k = idx / 512, n = idx % 512;
    out[bswz(k, n, 32)] = (_Float16)in[idx];
}

__global__ void conv_w1T(const float* __restrict__ w1, _Float16* __restrict__ out) { // [512 x 128] swizzled
    int idx = blockIdx.x * 256 + threadIdx.x;
    if (idx >= 512 * 128) return;
    int k = idx / 128, n = idx % 128;
    out[bswz(k, n, 8)] = (_Float16)w1[n * 512 + k];
}

__global__ void conv_bias_gi(const float* __restrict__ bf, const float* __restrict__ bb,
                             float* __restrict__ out) {          // [768]
    int i = blockIdx.x * 256 + threadIdx.x;
    if (i >= 768) return;
    out[i] = (i < 384) ? bf[i] : bb[i - 384];
}

// ---------------- stage 1: embedding gather + mean ----------------

__global__ __launch_bounds__(320) void embed_mean(const int* __restrict__ wid,
                                                  const float* __restrict__ tab,
                                                  _Float16* __restrict__ se) {
    int row = blockIdx.x;                 // b*S + s
    int e = threadIdx.x;
    __shared__ int ids[Wv];
    if (e < Wv) ids[e] = wid[(size_t)row * Wv + e];
    __syncthreads();
    float acc = 0.f;
    if (e < Ev) {
#pragma unroll 4
        for (int w = 0; w < Wv; ++w) {
            if (w + 1 < Wv) __builtin_prefetch(&tab[(size_t)ids[w + 1] * Ev + e], 0, 1);
            acc += tab[(size_t)ids[w] * Ev + e];
        }
    }
    if (e < KPAD)
        se[(size_t)row * KPAD + e] = (_Float16)((e < Ev) ? acc * (1.f / 32.f) : 0.f);
}

// ---------------- generic f16 WMMA GEMM ----------------
// C = act(A@B + bias).  A:[M,K] f16 row-major, Bsw: swizzled [K,N] f16, C:[M,N] f32.
// 128-thread blocks: 4 waves x 1 N-tile, each wave 2 M-tiles (B-fragment reuse x2).
// grid = (N/64, M/32).   ACT: 0=none, 1=relu, 2=tanh.

template <int ACT>
__global__ __launch_bounds__(128) void gemm_wmma(const _Float16* __restrict__ A,
                                                 const _Float16* __restrict__ Bsw,
                                                 const float* __restrict__ bias,
                                                 float* __restrict__ C,
                                                 int N, int K) {
    int wid = threadIdx.x >> 5;
    int nt  = blockIdx.x * 4 + wid;
    int mt0 = blockIdx.y * 2;
    int ntiles = N >> 4;
    const _Float16* A0 = A + (size_t)mt0 * 16 * K;
    const _Float16* A1 = A0 + (size_t)16 * K;
    fvec8 c0 = {0.f, 0.f, 0.f, 0.f, 0.f, 0.f, 0.f, 0.f};
    fvec8 c1 = {0.f, 0.f, 0.f, 0.f, 0.f, 0.f, 0.f, 0.f};
    int ktiles = K >> 5;
    for (int kt = 0; kt < ktiles; ++kt) {
        hvec16 b  = load_b_frag_sw(Bsw, kt, nt, ntiles);
        hvec16 a0 = load_a_frag(A0 + kt * 32, K);
        hvec16 a1 = load_a_frag(A1 + kt * 32, K);
        c0 = __builtin_amdgcn_wmma_f32_16x16x32_f16(false, a0, false, b, (short)0, c0, false, false);
        c1 = __builtin_amdgcn_wmma_f32_16x16x32_f16(false, a1, false, b, (short)0, c1, false, false);
    }
    int lane = threadIdx.x & 31;
    int n  = nt * 16 + (lane & 15);
    int mb0 = mt0 * 16 + ((lane >> 4) << 3);
    float bv = bias ? bias[n] : 0.f;
#pragma unroll
    for (int r = 0; r < 8; ++r) {
        float v0 = c0[r] + bv;
        float v1 = c1[r] + bv;
        if (ACT == 1) { v0 = v0 > 0.f ? v0 : 0.f; v1 = v1 > 0.f ? v1 : 0.f; }
        if (ACT == 2) { v0 = tanhf(v0); v1 = tanhf(v1); }
        C[(size_t)(mb0 + r) * N + n] = v0;
        C[(size_t)(mb0 + 16 + r) * N + n] = v1;
    }
}

// ---------------- stage 3: bidirectional GRU scan (1 block / direction, 16 waves) -----------

#define HS 136   // padded f16 row stride for h in LDS

__global__ __launch_bounds__(512) void gru_scan(const float* __restrict__ gi,        // [8192 x 768]
                                                const _Float16* __restrict__ WhhTsw, // [128 x 768] swizzled
                                                const float* __restrict__ bhhf,
                                                const float* __restrict__ bhhb,
                                                float* __restrict__ sent_rep,        // [8192 x 256]
                                                float* __restrict__ hT) {            // [2 x 32 x 128]
    int dir = blockIdx.x;
    const float* bhh = dir ? bhhb : bhhf;
    __shared__ _Float16 hH[32 * HS];      // h as f16 (WMMA A source)
    __shared__ float    gh[32 * 384];     // h @ Whh^T for this direction
    __shared__ float    hF[32 * 128];     // h master copy (f32)
    int tid = threadIdx.x;
    int wid = tid >> 5;

    // 48 (mtile,ntile) pairs over 16 waves, 3 each; B fragments resident in VGPRs.
    int pmt[3], pnt[3];
    hvec16 bf[3][4];
#pragma unroll
    for (int j = 0; j < 3; ++j) {
        int p = wid * 3 + j;
        pmt[j] = p / 24;
        pnt[j] = p % 24;
#pragma unroll
        for (int kt = 0; kt < 4; ++kt)
            bf[j][kt] = load_b_frag_sw(WhhTsw, kt, dir * 24 + pnt[j], 48);
    }
    for (int r = tid; r < 32 * 128; r += 512) hF[r] = 0.f;
    for (int r = tid; r < 32 * HS;  r += 512) hH[r] = (_Float16)0.f;
    __syncthreads();

    for (int t = 0; t < Sv; ++t) {
        // gh = h @ Whh^T  (24 N-tiles x 2 M-tiles, K=128 in 4 steps)
#pragma unroll
        for (int j = 0; j < 3; ++j) {
            hvec16 a[4];
#pragma unroll
            for (int kt = 0; kt < 4; ++kt)
                a[kt] = load_a_frag(hH + pmt[j] * 16 * HS + kt * 32, HS);
            fvec8 c = {0.f, 0.f, 0.f, 0.f, 0.f, 0.f, 0.f, 0.f};
#pragma unroll
            for (int kt = 0; kt < 4; ++kt)
                c = __builtin_amdgcn_wmma_f32_16x16x32_f16(false, a[kt], false, bf[j][kt],
                                                           (short)0, c, false, false);
            int lane = tid & 31;
            int n  = pnt[j] * 16 + (lane & 15);
            int mb = pmt[j] * 16 + ((lane >> 4) << 3);
#pragma unroll
            for (int r = 0; r < 8; ++r) gh[(mb + r) * 384 + n] = c[r];
        }
        __syncthreads();

        int srow = dir ? (Sv - 1 - t) : t;
#pragma unroll
        for (int r = 0; r < 8; ++r) {
            int i = tid + (r << 9);              // 0..4095
            int b = i >> 7, col = i & 127;
            size_t gib = ((size_t)(b * Sv + srow)) * 768 + dir * 384;
            float ir  = gi[gib + col];
            float iz  = gi[gib + 128 + col];
            float inn = gi[gib + 256 + col];
            float hr = gh[b * 384 + col]       + bhh[col];
            float hz = gh[b * 384 + 128 + col] + bhh[128 + col];
            float hn = gh[b * 384 + 256 + col] + bhh[256 + col];
            float rr = 1.f / (1.f + __expf(-(ir + hr)));
            float zz = 1.f / (1.f + __expf(-(iz + hz)));
            float nn = tanhf(inn + rr * hn);
            float hp = hF[i];
            float hnew = (1.f - zz) * nn + zz * hp;
            hF[i] = hnew;
            hH[b * HS + col] = (_Float16)hnew;
            sent_rep[((size_t)(b * Sv + srow)) * 256 + dir * 128 + col] = hnew;
        }
        __syncthreads();
    }
#pragma unroll
    for (int r = 0; r < 8; ++r) {
        int i = tid + (r << 9);
        hT[dir * 4096 + i] = hF[i];
    }
}

// ---------------- topic segment sums ----------------

__device__ __forceinline__ float padv(const float* sr, int b, int i, int c) {
    return (i >= 1 && i <= Sv) ? sr[((size_t)(b * Sv + (i - 1))) * 256 + c] : 0.f;
}

__global__ __launch_bounds__(256) void topic_mat_kernel(const float* __restrict__ sent_rep,
                                                        const int* __restrict__ tse,
                                                        float* __restrict__ topic_mat) {
    int bt = blockIdx.x;                 // b*16 + t
    int c  = threadIdx.x;                // 0..255
    int st = tse[bt * 2], en = tse[bt * 2 + 1];
    float v;
    if (c < 128) v = padv(sent_rep, bt >> 4, en, c) - padv(sent_rep, bt >> 4, st - 1, c);
    else         v = padv(sent_rep, bt >> 4, st, c) - padv(sent_rep, bt >> 4, en + 1, c);
    topic_mat[(size_t)bt * 256 + c] = v;
}

// doc_vec replicates the reference's stack([hT_f,hT_b],0).reshape(B,2H) flattening quirk.
__device__ __forceinline__ float doc_vec(const float* hT, int b, int c) {
    const float* src = hT + ((b < 16) ? 0 : 4096);
    return src[(((b & 15) * 2 + (c >> 7)) << 7) + (c & 127)];
}

__global__ __launch_bounds__(256) void build_cat(const float* __restrict__ sent_rep,
                                                 const float* __restrict__ hT,
                                                 const float* __restrict__ topic_mat,
                                                 const int* __restrict__ tse,
                                                 _Float16* __restrict__ catD,
                                                 _Float16* __restrict__ catT,
                                                 _Float16* __restrict__ inp_h) {
    int row = blockIdx.x;                // b*S + s
    int c = threadIdx.x;
    int b = row >> 8, s = row & 255;
    float sr = sent_rep[(size_t)row * 256 + c];
    float dv = doc_vec(hT, b, c);
    int cnt = 0;
#pragma unroll
    for (int t = 0; t < Tv; ++t) {
        int st = tse[(b * Tv + t) * 2];
        cnt += (s >= st - 1) ? 1 : 0;
    }
    int tid2 = cnt - 1;
    tid2 = tid2 < 0 ? 0 : (tid2 > Tv - 1 ? Tv - 1 : tid2);
    float tr = topic_mat[((size_t)(b * Tv + tid2)) * 256 + c];
    size_t o = (size_t)row * 512;
    catD[o + c]       = (_Float16)dv;
    catD[o + 256 + c] = (_Float16)sr;
    catT[o + c]       = (_Float16)tr;
    catT[o + 256 + c] = (_Float16)sr;
    inp_h[o + c]      = (_Float16)sr;
}

// ---------------- attention scores + softmax + context ----------------

__global__ __launch_bounds__(256) void score_reduce(const float* __restrict__ hta,
                                                    const float* __restrict__ v,
                                                    float* __restrict__ scores) {
    int row = blockIdx.x * 256 + threadIdx.x;   // 0..8191
    float acc = 0.f;
    for (int n = 0; n < 512; ++n) acc += hta[(size_t)row * 512 + n] * v[n];
    scores[row] = acc;
}

__device__ __forceinline__ float blk_max(float v, float* red) {
    int t = threadIdx.x;
    red[t] = v; __syncthreads();
    for (int o = 128; o > 0; o >>= 1) { if (t < o) red[t] = fmaxf(red[t], red[t + o]); __syncthreads(); }
    float r = red[0]; __syncthreads(); return r;
}
__device__ __forceinline__ float blk_sum(float v, float* red) {
    int t = threadIdx.x;
    red[t] = v; __syncthreads();
    for (int o = 128; o > 0; o >>= 1) { if (t < o) red[t] += red[t + o]; __syncthreads(); }
    float r = red[0]; __syncthreads(); return r;
}

__global__ __launch_bounds__(256) void softmax_context(const float* __restrict__ scores,
                                                       const float* __restrict__ hT,
                                                       const float* __restrict__ topic_mat,
                                                       const int* __restrict__ tse,
                                                       _Float16* __restrict__ inp_h) {
    int b = blockIdx.x, s = threadIdx.x;
    __shared__ float red[256];
    float ds = scores[b * 256 + s];
    float ts = scores[8192 + b * 256 + s];
    float m1 = blk_max(ds, red);
    float e1 = __expf(ds - m1);
    float s1 = blk_sum(e1, red);
    float m2 = blk_max(ts, red);
    float e2 = __expf(ts - m2);
    float s2 = blk_sum(e2, red);
    float dw = e1 / s1, tw = e2 / s2;
    int cnt = 0;
#pragma unroll
    for (int t = 0; t < Tv; ++t) {
        int st = tse[(b * Tv + t) * 2];
        cnt += (s >= st - 1) ? 1 : 0;
    }
    int tid2 = cnt - 1;
    tid2 = tid2 < 0 ? 0 : (tid2 > Tv - 1 ? Tv - 1 : tid2);
    size_t row = (size_t)(b * Sv + s);
    for (int c = 0; c < 256; ++c) {
        float ctx = dw * doc_vec(hT, b, c)
                  + tw * topic_mat[((size_t)(b * Tv + tid2)) * 256 + c];
        inp_h[row * 512 + 256 + c] = (_Float16)ctx;
    }
}

// ---------------- final projection ----------------

__global__ __launch_bounds__(256) void final_logits(const float* __restrict__ hdd,
                                                    const float* __restrict__ W2,
                                                    const float* __restrict__ b2,
                                                    float* __restrict__ out) {
    int row = blockIdx.x * 256 + threadIdx.x;   // 0..8191
    float acc = b2[0];
#pragma unroll 8
    for (int j = 0; j < 128; ++j) acc += hdd[(size_t)row * 128 + j] * W2[j];
    out[row] = acc;
}

// ---------------- host launch ----------------

extern "C" void kernel_launch(void* const* d_in, const int* in_sizes, int n_in,
                              void* d_out, int out_size, void* d_ws, size_t ws_size,
                              hipStream_t stream) {
    const int*   word_ids = (const int*)d_in[0];
    const int*   tse      = (const int*)d_in[1];
    const float* table    = (const float*)d_in[2];
    const float* Wih_f    = (const float*)d_in[3];
    const float* Whh_f    = (const float*)d_in[4];
    const float* bih_f    = (const float*)d_in[5];
    const float* bhh_f    = (const float*)d_in[6];
    const float* Wih_b    = (const float*)d_in[7];
    const float* Whh_b    = (const float*)d_in[8];
    const float* bih_b    = (const float*)d_in[9];
    const float* bhh_b    = (const float*)d_in[10];
    const float* v_att    = (const float*)d_in[11];
    const float* W_att    = (const float*)d_in[12];
    const float* W1       = (const float*)d_in[13];
    const float* b1       = (const float*)d_in[14];
    const float* W2       = (const float*)d_in[15];
    const float* b2       = (const float*)d_in[16];
    float* logits = (float*)d_out;

    char* w = (char*)d_ws;
    size_t off = 0;
    auto carve = [&](size_t bytes) -> void* {
        void* p = w + off;
        off = (off + bytes + 255) & ~(size_t)255;
        return p;
    };
    _Float16* sent_emb_h = (_Float16*)carve((size_t)Mrows * KPAD * 2);   // 5.2 MB
    _Float16* giB_h      = (_Float16*)carve((size_t)KPAD * 768 * 2);     // swizzled
    float*    bias_gi    = (float*)   carve(768 * 4);
    float*    gi         = (float*)   carve((size_t)Mrows * 768 * 4);    // 25 MB
    _Float16* WhhT_sw    = (_Float16*)carve((size_t)Hv * 768 * 2);       // swizzled
    float*    sent_rep   = (float*)   carve((size_t)Mrows * 256 * 4);    // 8.4 MB
    float*    hT         = (float*)   carve(2 * 32 * 128 * 4);
    float*    topic_mat  = (float*)   carve((size_t)Bv * Tv * 256 * 4);
    _Float16* Watt_sw    = (_Float16*)carve((size_t)512 * 512 * 2);      // swizzled
    _Float16* W1T_sw     = (_Float16*)carve((size_t)512 * 128 * 2);      // swizzled
    _Float16* catD_h     = (_Float16*)carve((size_t)Mrows * 512 * 2);    // 8.4 MB
    _Float16* catT_h     = (_Float16*)carve((size_t)Mrows * 512 * 2);
    _Float16* inp_h      = (_Float16*)carve((size_t)Mrows * 512 * 2);
    float*    hta        = (float*)   carve((size_t)Mrows * 512 * 4);    // 16.8 MB (reused)
    float*    scores     = (float*)   carve(2 * Mrows * 4);
    float*    hdd        = (float*)   carve((size_t)Mrows * 128 * 4);

    // weight prep (f32 -> f16, fragment-swizzled B layouts)
    conv_giB  <<<(KPAD * 768 + 255) / 256, 256, 0, stream>>>(Wih_f, Wih_b, giB_h);
    conv_whhT <<<(Hv * 768 + 255) / 256,   256, 0, stream>>>(Whh_f, Whh_b, WhhT_sw);
    conv_watt <<<(512 * 512 + 255) / 256,  256, 0, stream>>>(W_att, Watt_sw);
    conv_w1T  <<<(512 * 128 + 255) / 256,  256, 0, stream>>>(W1, W1T_sw);
    conv_bias_gi<<<3, 256, 0, stream>>>(bih_f, bih_b, bias_gi);

    // 1) embedding gather + mean  (HBM-bandwidth stage)
    embed_mean<<<Mrows, 320, 0, stream>>>(word_ids, table, sent_emb_h);

    // 2) gi = sent_emb @ [Wih_f ; Wih_b]^T + bias   (M=8192, N=768, K=320)
    gemm_wmma<0><<<dim3(768 / 64, Mrows / 32), 128, 0, stream>>>(sent_emb_h, giB_h, bias_gi, gi, 768, KPAD);

    // 3) bidirectional GRU scan (sequential; LDS/VGPR resident)
    gru_scan<<<2, 512, 0, stream>>>(gi, WhhT_sw, bhh_f, bhh_b, sent_rep, hT);

    // 4) topic segment vectors, cat matrices
    topic_mat_kernel<<<Bv * Tv, 256, 0, stream>>>(sent_rep, tse, topic_mat);
    build_cat<<<Mrows, 256, 0, stream>>>(sent_rep, hT, topic_mat, tse, catD_h, catT_h, inp_h);

    // 5) attention: tanh(cat @ W_att) then dot with v  (M=8192, N=K=512); hta reused
    gemm_wmma<2><<<dim3(512 / 64, Mrows / 32), 128, 0, stream>>>(catD_h, Watt_sw, nullptr, hta, 512, 512);
    score_reduce<<<Mrows / 256, 256, 0, stream>>>(hta, v_att, scores);
    gemm_wmma<2><<<dim3(512 / 64, Mrows / 32), 128, 0, stream>>>(catT_h, Watt_sw, nullptr, hta, 512, 512);
    score_reduce<<<Mrows / 256, 256, 0, stream>>>(hta, v_att, scores + Mrows);

    // 6) softmax + context -> inp[:,256:512]
    softmax_context<<<Bv, 256, 0, stream>>>(scores, hT, topic_mat, tse, inp_h);

    // 7) MLP: hdd = relu(inp @ W1^T + b1)  (M=8192, N=128, K=512)
    gemm_wmma<1><<<dim3(128 / 64, Mrows / 32), 128, 0, stream>>>(inp_h, W1T_sw, b1, hdd, 128, 512);

    // 8) logits = hdd @ W2^T + b2
    final_logits<<<Mrows / 256, 256, 0, stream>>>(hdd, W2, b2, logits);
}
